// EncLayer_3504693314244
// MI455X (gfx1250) — compile-verified, matrix operand order checked by reading, not run on hardware
//
#include <hip/hip_runtime.h>
#include <hip/hip_bf16.h>
#include <math.h>

// ---------------------------------------------------------------------------
// EncLayer (ProteinMPNN-style) for MI455X / gfx1250.
// B=2, L=2048, K=48, H=128, din_msg=384.  wave32, WMMA bf16 16x16x32.
// Register-blocked: one N-tile per wave, 6 M-tiles per wave, B-fragments
// loaded once per kt and reused across all M-tiles.
// ---------------------------------------------------------------------------

#define Bdim 2
#define Ldim 2048
#define Kn   48
#define Hd   128

#define NODES1 2            // nodes per workgroup in edge-MLP kernels
#define R1     (NODES1*Kn)  // 96 edge rows per WG -> 6 M-tiles
#define SA     392          // LDS stride for 384-wide activations (bf16, +8 pad)
#define S1     136          // LDS stride for 128-wide activations (+8 pad)
#define SH     520          // LDS stride for 512-wide FFN hidden (+8 pad)
#define N2     16           // nodes per workgroup in FFN kernel

typedef __attribute__((ext_vector_type(16))) __bf16 v16bf;
typedef __attribute__((ext_vector_type(8)))  float  v8f;
typedef __attribute__((ext_vector_type(8)))  unsigned short u16x8;

union BF16x16 { v16bf v; u16x8 h[2]; };

// ---- WMMA helpers ---------------------------------------------------------

__device__ inline v8f wmma_bf16(v16bf a, v16bf b, v8f c) {
  // D = A(16x32 bf16) x B(32x16 bf16) + C(16x16 f32)
  return __builtin_amdgcn_wmma_f32_16x16x32_bf16(
      /*neg_a=*/false, a, /*neg_b=*/false, b,
      /*c_mod=*/(short)0, c, /*reuse_a=*/false, /*reuse_b=*/false);
}

// A fragment from LDS (row-major bf16 activations, padded stride).
// ISA 16-bit A 16x32 layout: lanes 0-15 -> M=lane, elems 0..7 = K 0..7,
// elems 8..15 = K 16..23; lanes 16-31 -> M=lane-16, K ranges shifted by +8.
__device__ inline v16bf lds_a_frag(const __bf16* base, int stride,
                                   int mtile, int kt, int lane) {
  int row  = mtile * 16 + (lane & 15);
  int half = lane >> 4;
  const __bf16* p = base + row * stride + kt * 32 + half * 8;
  BF16x16 u;
  u.h[0] = *(const u16x8*)(p);        // K block lo (8 bf16, 16B aligned)
  u.h[1] = *(const u16x8*)(p + 16);   // K block hi
  return u.v;
}

// B fragment from global: weights pre-packed fragment-major by pack_w,
// each lane reads 16 contiguous bf16 (32B), 1KB per wave per fragment.
__device__ inline v16bf glb_b_frag(const __bf16* __restrict__ Wp, int NT,
                                   int kt, int nt, int lane) {
  const __bf16* p = Wp + ((size_t)((kt * NT + nt) * 32 + lane) << 4);
  BF16x16 u;
  u.h[0] = *(const u16x8*)(p);
  u.h[1] = *(const u16x8*)(p + 8);
  return u.v;
}

__device__ inline float gelu_exact(float x) {
  return 0.5f * x * (1.0f + erff(x * 0.70710678118654752f));
}

// ---- weight pre-pack: f32 row-major [din,dout] -> bf16 B-fragments --------
// Fragment (kt,nt) holds B tile K=[kt*32,kt*32+32), N=[nt*16,nt*16+16).
// Per-lane element e maps to K = kt*32 + e + (e&8) + (lane>=16 ? 8 : 0),
// N = nt*16 + (lane&15)  (mirror of the dense 16-bit A layout).
__global__ __launch_bounds__(256) void pack_w(const float* __restrict__ W,
                                              __bf16* __restrict__ dst,
                                              int din, int dout) {
  int total = din * dout;
  int NT = dout >> 4;
  for (int idx = blockIdx.x * blockDim.x + threadIdx.x; idx < total;
       idx += gridDim.x * blockDim.x) {
    int frag = idx >> 9;          // 512 elems per fragment
    int r    = idx & 511;
    int lane = r >> 4, e = r & 15;
    int kt = frag / NT, nt = frag - kt * NT;
    int k = kt * 32 + e + (e & 8) + ((lane >> 4) << 3);
    int n = nt * 16 + (lane & 15);
    dst[idx] = (__bf16)W[k * dout + n];
  }
}

// ---- kernel 1: message MLP + masked K-sum + residual LayerNorm ------------
__global__ __launch_bounds__(256) void k_msg_node(
    const float* __restrict__ hV, const float* __restrict__ hE,
    const int* __restrict__ Eidx, const float* __restrict__ maskA,
    const __bf16* __restrict__ W1p, const float* __restrict__ b1,
    const __bf16* __restrict__ W2p, const float* __restrict__ b2,
    const __bf16* __restrict__ W3p, const float* __restrict__ b3,
    const float* __restrict__ g1, const float* __restrict__ be1,
    float* __restrict__ hVo) {
  extern __shared__ char smem[];
  __bf16* XA   = (__bf16*)smem;                           // 96*392*2 = 75264B
  __bf16* O1   = (__bf16*)(smem + 75264);                 // 96*136*2 = 26112B
  __bf16* O2   = (__bf16*)(smem + 75264 + 26112);         // 26112B
  float*  mbuf = (float*)(smem + 75264 + 2 * 26112);      // 384B
  float*  dh   = (float*)(smem + 75264 + 2 * 26112 + 384);// 1024B

  int tid = threadIdx.x, lane = tid & 31, wid = tid >> 5;
  int half = lane >> 4, cidx = lane & 15;
  int nodeBase = blockIdx.x * NODES1;

  for (int r = tid; r < R1; r += 256) mbuf[r] = maskA[nodeBase * Kn + r];
  // stage XA = [h_V(center) | h_E | h_V[E_idx]] as bf16
  for (int idx = tid; idx < R1 * 384; idx += 256) {
    int r = idx / 384, c = idx - r * 384;
    int node = nodeBase + (r / Kn);
    int kk   = r % Kn;
    float v;
    if (c < Hd) {
      v = hV[node * Hd + c];
    } else if (c < 2 * Hd) {
      v = hE[(size_t)(node * Kn + kk) * Hd + (c - Hd)];
    } else {
      int bb = node / Ldim;
      int j  = Eidx[node * Kn + kk];
      v = hV[(size_t)(bb * Ldim + j) * Hd + (c - 2 * Hd)];
    }
    XA[r * SA + c] = (__bf16)v;
  }
  __syncthreads();

  int nt = wid;  // each of the 8 waves owns one 16-col N-tile

  // layer 1: [96x384] @ [384x128]; B frags reused across 6 M-tiles
  {
    float bb = b1[nt * 16 + cidx];
    v8f acc[6];
#pragma unroll
    for (int mt = 0; mt < 6; ++mt)
      for (int j = 0; j < 8; ++j) acc[mt][j] = bb;
#pragma unroll
    for (int kc = 0; kc < 12; kc += 4) {
      v16bf B0 = glb_b_frag(W1p, 8, kc + 0, nt, lane);
      v16bf B1 = glb_b_frag(W1p, 8, kc + 1, nt, lane);
      v16bf B2 = glb_b_frag(W1p, 8, kc + 2, nt, lane);
      v16bf B3 = glb_b_frag(W1p, 8, kc + 3, nt, lane);
#pragma unroll
      for (int mt = 0; mt < 6; ++mt) {
        acc[mt] = wmma_bf16(lds_a_frag(XA, SA, mt, kc + 0, lane), B0, acc[mt]);
        acc[mt] = wmma_bf16(lds_a_frag(XA, SA, mt, kc + 1, lane), B1, acc[mt]);
        acc[mt] = wmma_bf16(lds_a_frag(XA, SA, mt, kc + 2, lane), B2, acc[mt]);
        acc[mt] = wmma_bf16(lds_a_frag(XA, SA, mt, kc + 3, lane), B3, acc[mt]);
      }
    }
#pragma unroll
    for (int mt = 0; mt < 6; ++mt)
      for (int j = 0; j < 8; ++j)
        O1[(mt * 16 + half * 8 + j) * S1 + nt * 16 + cidx] =
            (__bf16)gelu_exact(acc[mt][j]);
  }
  __syncthreads();

  // layer 2: [96x128] @ [128x128]; 4 B frags preloaded
  {
    float bb = b2[nt * 16 + cidx];
    v16bf B0 = glb_b_frag(W2p, 8, 0, nt, lane);
    v16bf B1 = glb_b_frag(W2p, 8, 1, nt, lane);
    v16bf B2 = glb_b_frag(W2p, 8, 2, nt, lane);
    v16bf B3 = glb_b_frag(W2p, 8, 3, nt, lane);
#pragma unroll
    for (int mt = 0; mt < 6; ++mt) {
      v8f c; for (int j = 0; j < 8; ++j) c[j] = bb;
      c = wmma_bf16(lds_a_frag(O1, S1, mt, 0, lane), B0, c);
      c = wmma_bf16(lds_a_frag(O1, S1, mt, 1, lane), B1, c);
      c = wmma_bf16(lds_a_frag(O1, S1, mt, 2, lane), B2, c);
      c = wmma_bf16(lds_a_frag(O1, S1, mt, 3, lane), B3, c);
#pragma unroll
      for (int j = 0; j < 8; ++j)
        O2[(mt * 16 + half * 8 + j) * S1 + nt * 16 + cidx] =
            (__bf16)gelu_exact(c[j]);
    }
  }
  __syncthreads();

  // layer 3 + masked reduction over K; wave owns disjoint 16 cols of dh,
  // so per-node sums accumulate in registers (no LDS atomics needed).
  {
    float bb = b3[nt * 16 + cidx];
    v16bf B0 = glb_b_frag(W3p, 8, 0, nt, lane);
    v16bf B1 = glb_b_frag(W3p, 8, 1, nt, lane);
    v16bf B2 = glb_b_frag(W3p, 8, 2, nt, lane);
    v16bf B3 = glb_b_frag(W3p, 8, 3, nt, lane);
    float sn0 = 0.0f, sn1 = 0.0f;
#pragma unroll
    for (int mt = 0; mt < 6; ++mt) {
      v8f c; for (int j = 0; j < 8; ++j) c[j] = bb;
      c = wmma_bf16(lds_a_frag(O2, S1, mt, 0, lane), B0, c);
      c = wmma_bf16(lds_a_frag(O2, S1, mt, 1, lane), B1, c);
      c = wmma_bf16(lds_a_frag(O2, S1, mt, 2, lane), B2, c);
      c = wmma_bf16(lds_a_frag(O2, S1, mt, 3, lane), B3, c);
      float s = 0.0f;
#pragma unroll
      for (int j = 0; j < 8; ++j) s += c[j] * mbuf[mt * 16 + half * 8 + j];
      s += __shfl_xor(s, 16, 32);    // combine both row halves
      if (mt < 3) sn0 += s; else sn1 += s;   // 3 M-tiles per node
    }
    if (lane < 16) {
      dh[nt * 16 + lane]      = sn0;
      dh[Hd + nt * 16 + lane] = sn1;
    }
  }
  __syncthreads();

  // residual + LayerNorm, one wave per node
  if (wid < NODES1) {
    int node = nodeBase + wid;
    float x[4], sum = 0.0f;
#pragma unroll
    for (int i = 0; i < 4; ++i) {
      int col = lane + 32 * i;
      x[i] = hV[node * Hd + col] + dh[wid * Hd + col] * (1.0f / 30.0f);
      sum += x[i];
    }
#pragma unroll
    for (int off = 16; off; off >>= 1) sum += __shfl_xor(sum, off, 32);
    float mu = sum * (1.0f / Hd);
    float var = 0.0f;
#pragma unroll
    for (int i = 0; i < 4; ++i) { float d = x[i] - mu; var += d * d; }
#pragma unroll
    for (int off = 16; off; off >>= 1) var += __shfl_xor(var, off, 32);
    float rstd = rsqrtf(var * (1.0f / Hd) + 1e-5f);
#pragma unroll
    for (int i = 0; i < 4; ++i) {
      int col = lane + 32 * i;
      hVo[node * Hd + col] = (x[i] - mu) * rstd * g1[col] + be1[col];
    }
  }
}

// ---- kernel 2: position-wise FFN + residual LN + node mask (in place) -----
__global__ __launch_bounds__(256) void k_ffn(
    float* __restrict__ hVio, const float* __restrict__ maskV,
    const __bf16* __restrict__ Wip, const float* __restrict__ bi,
    const __bf16* __restrict__ Wop, const float* __restrict__ bo,
    const float* __restrict__ g2, const float* __restrict__ be2) {
  extern __shared__ char smem[];
  __bf16* X  = (__bf16*)smem;                   // 16*136*2 = 4352B
  __bf16* Hh = (__bf16*)(smem + 4352);          // 16*520*2 = 16640B
  float*  dh = (float*)(smem + 4352 + 16640);   // 16*128*4 = 8192B

  int tid = threadIdx.x, lane = tid & 31, wid = tid >> 5;
  int half = lane >> 4, cidx = lane & 15;
  int nodeBase = blockIdx.x * N2;

  for (int idx = tid; idx < N2 * Hd; idx += 256) {
    int r = idx >> 7, c = idx & 127;
    X[r * S1 + c] = (__bf16)hVio[(size_t)(nodeBase + r) * Hd + c];
  }
  __syncthreads();

  // GEMM1: [16x128] @ [128x512] -> GELU -> Hh; A frags shared by 4 N-tiles
  {
    v16bf A0 = lds_a_frag(X, S1, 0, 0, lane);
    v16bf A1 = lds_a_frag(X, S1, 0, 1, lane);
    v16bf A2 = lds_a_frag(X, S1, 0, 2, lane);
    v16bf A3 = lds_a_frag(X, S1, 0, 3, lane);
#pragma unroll
    for (int q = 0; q < 4; ++q) {
      int nt = wid * 4 + q;
      float bb = bi[nt * 16 + cidx];
      v8f c; for (int j = 0; j < 8; ++j) c[j] = bb;
      c = wmma_bf16(A0, glb_b_frag(Wip, 32, 0, nt, lane), c);
      c = wmma_bf16(A1, glb_b_frag(Wip, 32, 1, nt, lane), c);
      c = wmma_bf16(A2, glb_b_frag(Wip, 32, 2, nt, lane), c);
      c = wmma_bf16(A3, glb_b_frag(Wip, 32, 3, nt, lane), c);
#pragma unroll
      for (int j = 0; j < 8; ++j)
        Hh[(half * 8 + j) * SH + nt * 16 + cidx] = (__bf16)gelu_exact(c[j]);
    }
  }
  __syncthreads();

  // GEMM2: [16x512] @ [512x128] -> dh; 8 ntiles, 1/wave, 16 ktiles
  {
    int nt = wid;
    float bb = bo[nt * 16 + cidx];
    v8f c; for (int j = 0; j < 8; ++j) c[j] = bb;
#pragma unroll
    for (int kt = 0; kt < 16; ++kt)
      c = wmma_bf16(lds_a_frag(Hh, SH, 0, kt, lane),
                    glb_b_frag(Wop, 8, kt, nt, lane), c);
#pragma unroll
    for (int j = 0; j < 8; ++j)
      dh[(half * 8 + j) * Hd + nt * 16 + cidx] = c[j];
  }
  __syncthreads();

  // residual + LN + mask_V; wave handles 2 nodes (reads f32 residual first)
  for (int ni = 0; ni < 2; ++ni) {
    int nl = wid * 2 + ni, node = nodeBase + nl;
    float x[4], sum = 0.0f;
#pragma unroll
    for (int i = 0; i < 4; ++i) {
      int col = lane + 32 * i;
      x[i] = hVio[node * Hd + col] + dh[nl * Hd + col];
      sum += x[i];
    }
#pragma unroll
    for (int off = 16; off; off >>= 1) sum += __shfl_xor(sum, off, 32);
    float mu = sum * (1.0f / Hd);
    float var = 0.0f;
#pragma unroll
    for (int i = 0; i < 4; ++i) { float d = x[i] - mu; var += d * d; }
#pragma unroll
    for (int off = 16; off; off >>= 1) var += __shfl_xor(var, off, 32);
    float rstd = rsqrtf(var * (1.0f / Hd) + 1e-5f);
    float m = maskV[node];
#pragma unroll
    for (int i = 0; i < 4; ++i) {
      int col = lane + 32 * i;
      hVio[node * Hd + col] = m * ((x[i] - mu) * rstd * g2[col] + be2[col]);
    }
  }
}

// ---- kernel 3: edge-update MLP + per-edge residual LayerNorm --------------
__global__ __launch_bounds__(256) void k_edge(
    const float* __restrict__ hV,  // updated h_V (after FFN/mask)
    const float* __restrict__ hE, const int* __restrict__ Eidx,
    const __bf16* __restrict__ W1p, const float* __restrict__ b1,
    const __bf16* __restrict__ W2p, const float* __restrict__ b2,
    const __bf16* __restrict__ W3p, const float* __restrict__ b3,
    const float* __restrict__ g3, const float* __restrict__ be3,
    float* __restrict__ hEo) {
  extern __shared__ char smem[];
  __bf16* XA = (__bf16*)smem;                     // 75264B
  __bf16* O1 = (__bf16*)(smem + 75264);           // 26112B
  __bf16* O2 = (__bf16*)(smem + 75264 + 26112);   // 26112B
  float*  MSG = (float*)smem;  // reuses XA region after layer1 (52224B<=75264)

  int tid = threadIdx.x, lane = tid & 31, wid = tid >> 5;
  int half = lane >> 4, cidx = lane & 15;
  int nodeBase = blockIdx.x * NODES1;

  for (int idx = tid; idx < R1 * 384; idx += 256) {
    int r = idx / 384, c = idx - r * 384;
    int node = nodeBase + (r / Kn);
    int kk   = r % Kn;
    float v;
    if (c < Hd) {
      v = hV[node * Hd + c];
    } else if (c < 2 * Hd) {
      v = hE[(size_t)(node * Kn + kk) * Hd + (c - Hd)];
    } else {
      int bb = node / Ldim;
      int j  = Eidx[node * Kn + kk];
      v = hV[(size_t)(bb * Ldim + j) * Hd + (c - 2 * Hd)];
    }
    XA[r * SA + c] = (__bf16)v;
  }
  __syncthreads();

  int nt = wid;

  // layer 1
  {
    float bb = b1[nt * 16 + cidx];
    v8f acc[6];
#pragma unroll
    for (int mt = 0; mt < 6; ++mt)
      for (int j = 0; j < 8; ++j) acc[mt][j] = bb;
#pragma unroll
    for (int kc = 0; kc < 12; kc += 4) {
      v16bf B0 = glb_b_frag(W1p, 8, kc + 0, nt, lane);
      v16bf B1 = glb_b_frag(W1p, 8, kc + 1, nt, lane);
      v16bf B2 = glb_b_frag(W1p, 8, kc + 2, nt, lane);
      v16bf B3 = glb_b_frag(W1p, 8, kc + 3, nt, lane);
#pragma unroll
      for (int mt = 0; mt < 6; ++mt) {
        acc[mt] = wmma_bf16(lds_a_frag(XA, SA, mt, kc + 0, lane), B0, acc[mt]);
        acc[mt] = wmma_bf16(lds_a_frag(XA, SA, mt, kc + 1, lane), B1, acc[mt]);
        acc[mt] = wmma_bf16(lds_a_frag(XA, SA, mt, kc + 2, lane), B2, acc[mt]);
        acc[mt] = wmma_bf16(lds_a_frag(XA, SA, mt, kc + 3, lane), B3, acc[mt]);
      }
    }
#pragma unroll
    for (int mt = 0; mt < 6; ++mt)
      for (int j = 0; j < 8; ++j)
        O1[(mt * 16 + half * 8 + j) * S1 + nt * 16 + cidx] =
            (__bf16)gelu_exact(acc[mt][j]);
  }
  __syncthreads();

  // layer 2
  {
    float bb = b2[nt * 16 + cidx];
    v16bf B0 = glb_b_frag(W2p, 8, 0, nt, lane);
    v16bf B1 = glb_b_frag(W2p, 8, 1, nt, lane);
    v16bf B2 = glb_b_frag(W2p, 8, 2, nt, lane);
    v16bf B3 = glb_b_frag(W2p, 8, 3, nt, lane);
#pragma unroll
    for (int mt = 0; mt < 6; ++mt) {
      v8f c; for (int j = 0; j < 8; ++j) c[j] = bb;
      c = wmma_bf16(lds_a_frag(O1, S1, mt, 0, lane), B0, c);
      c = wmma_bf16(lds_a_frag(O1, S1, mt, 1, lane), B1, c);
      c = wmma_bf16(lds_a_frag(O1, S1, mt, 2, lane), B2, c);
      c = wmma_bf16(lds_a_frag(O1, S1, mt, 3, lane), B3, c);
#pragma unroll
      for (int j = 0; j < 8; ++j)
        O2[(mt * 16 + half * 8 + j) * S1 + nt * 16 + cidx] =
            (__bf16)gelu_exact(c[j]);
    }
  }
  __syncthreads();

  // layer 3 -> MSG (f32, reuses XA region; safe: XA dead after layer 1 and
  // all waves passed the layer-2 barrier)
  {
    float bb = b3[nt * 16 + cidx];
    v16bf B0 = glb_b_frag(W3p, 8, 0, nt, lane);
    v16bf B1 = glb_b_frag(W3p, 8, 1, nt, lane);
    v16bf B2 = glb_b_frag(W3p, 8, 2, nt, lane);
    v16bf B3 = glb_b_frag(W3p, 8, 3, nt, lane);
#pragma unroll
    for (int mt = 0; mt < 6; ++mt) {
      v8f c; for (int j = 0; j < 8; ++j) c[j] = bb;
      c = wmma_bf16(lds_a_frag(O2, S1, mt, 0, lane), B0, c);
      c = wmma_bf16(lds_a_frag(O2, S1, mt, 1, lane), B1, c);
      c = wmma_bf16(lds_a_frag(O2, S1, mt, 2, lane), B2, c);
      c = wmma_bf16(lds_a_frag(O2, S1, mt, 3, lane), B3, c);
#pragma unroll
      for (int j = 0; j < 8; ++j)
        MSG[(mt * 16 + half * 8 + j) * S1 + nt * 16 + cidx] = c[j];
    }
  }
  __syncthreads();

  // per-edge-row residual LayerNorm: 8 waves x 12 rows
  for (int r = wid * 12; r < wid * 12 + 12; ++r) {
    size_t erow = (size_t)(nodeBase * Kn + r) * Hd;
    float x[4], sum = 0.0f;
#pragma unroll
    for (int i = 0; i < 4; ++i) {
      int col = lane + 32 * i;
      x[i] = hE[erow + col] + MSG[r * S1 + col];
      sum += x[i];
    }
#pragma unroll
    for (int off = 16; off; off >>= 1) sum += __shfl_xor(sum, off, 32);
    float mu = sum * (1.0f / Hd);
    float var = 0.0f;
#pragma unroll
    for (int i = 0; i < 4; ++i) { float d = x[i] - mu; var += d * d; }
#pragma unroll
    for (int off = 16; off; off >>= 1) var += __shfl_xor(var, off, 32);
    float rstd = rsqrtf(var * (1.0f / Hd) + 1e-5f);
#pragma unroll
    for (int i = 0; i < 4; ++i) {
      int col = lane + 32 * i;
      hEo[erow + col] = (x[i] - mu) * rstd * g3[col] + be3[col];
    }
  }
}

// ---------------------------------------------------------------------------

extern "C" void kernel_launch(void* const* d_in, const int* in_sizes, int n_in,
                              void* d_out, int out_size, void* d_ws,
                              size_t ws_size, hipStream_t stream) {
  (void)in_sizes; (void)n_in; (void)out_size; (void)ws_size;

  const float* hV    = (const float*)d_in[0];
  const float* hE    = (const float*)d_in[1];
  const int*   Eidx  = (const int*)d_in[2];
  const float* maskV = (const float*)d_in[3];
  const float* maskA = (const float*)d_in[4];
  const float* W1w  = (const float*)d_in[5];  const float* W1b  = (const float*)d_in[6];
  const float* W2w  = (const float*)d_in[7];  const float* W2b  = (const float*)d_in[8];
  const float* W3w  = (const float*)d_in[9];  const float* W3b  = (const float*)d_in[10];
  const float* W11w = (const float*)d_in[11]; const float* W11b = (const float*)d_in[12];
  const float* W12w = (const float*)d_in[13]; const float* W12b = (const float*)d_in[14];
  const float* W13w = (const float*)d_in[15]; const float* W13b = (const float*)d_in[16];
  const float* Wiw  = (const float*)d_in[17]; const float* Wib  = (const float*)d_in[18];
  const float* Wow  = (const float*)d_in[19]; const float* Wob  = (const float*)d_in[20];
  const float* g1 = (const float*)d_in[21]; const float* be1 = (const float*)d_in[22];
  const float* g2 = (const float*)d_in[23]; const float* be2 = (const float*)d_in[24];
  const float* g3 = (const float*)d_in[25]; const float* be3 = (const float*)d_in[26];

  float* hVo = (float*)d_out;                              // [B,L,H]
  float* hEo = hVo + (size_t)Bdim * Ldim * Hd;             // [B,L,K,H]

  // bf16 fragment-packed weights in workspace (576KB total)
  __bf16* W1p  = (__bf16*)d_ws;
  __bf16* W2p  = W1p  + 384 * 128;
  __bf16* W3p  = W2p  + 128 * 128;
  __bf16* W11p = W3p  + 128 * 128;
  __bf16* W12p = W11p + 384 * 128;
  __bf16* W13p = W12p + 128 * 128;
  __bf16* Wip  = W13p + 128 * 128;
  __bf16* Wop  = Wip  + 128 * 512;

  pack_w<<<192, 256, 0, stream>>>(W1w,  W1p,  384, 128);
  pack_w<<< 64, 256, 0, stream>>>(W2w,  W2p,  128, 128);
  pack_w<<< 64, 256, 0, stream>>>(W3w,  W3p,  128, 128);
  pack_w<<<192, 256, 0, stream>>>(W11w, W11p, 384, 128);
  pack_w<<< 64, 256, 0, stream>>>(W12w, W12p, 128, 128);
  pack_w<<< 64, 256, 0, stream>>>(W13w, W13p, 128, 128);
  pack_w<<<256, 256, 0, stream>>>(Wiw,  Wip,  128, 512);
  pack_w<<<256, 256, 0, stream>>>(Wow,  Wop,  512, 128);

  const int nBlocksEdge = (Bdim * Ldim) / NODES1;   // 2048
  const size_t smem1 = 75264 + 2 * 26112 + 384 + 1024;  // 128,896B
  const size_t smem3 = 75264 + 2 * 26112;               // 127,488B
  const size_t smem2 = 4352 + 16640 + 8192;             //  29,184B

  k_msg_node<<<nBlocksEdge, 256, smem1, stream>>>(
      hV, hE, Eidx, maskA, W1p, W1b, W2p, W2b, W3p, W3b, g1, be1, hVo);

  k_ffn<<<(Bdim * Ldim) / N2, 256, smem2, stream>>>(
      hVo, maskV, Wip, Wib, Wop, Wob, g2, be2);

  k_edge<<<nBlocksEdge, 256, smem3, stream>>>(
      hVo, hE, Eidx, W11p, W11b, W12p, W12b, W13p, W13b, g3, be3, hEo);
}
